// MultiHeadAttention_query_8100308321042
// MI455X (gfx1250) — compile-verified, matrix operand order checked by reading, not run on hardware
//
#include <hip/hip_runtime.h>
#include <hip/hip_bf16.h>
#include <math.h>

typedef __attribute__((ext_vector_type(16))) _Float16 v16h;
typedef __attribute__((ext_vector_type(8)))  _Float16 v8h;
typedef __attribute__((ext_vector_type(8)))  float    v8f;

// Problem constants (match setup_inputs: Q=262144, B=1, C=128, H=W=512, ws=4, nh=4)
#define Q_TOTAL 262144
#define L_TOTAL 262144
#define C_DIM   128
#define NH      4
#define HD      32

// Workspace layout (float offsets). Uses 6144 floats = 24 KB of d_ws.
#define WS_KSUM   0      // f32 [16][128]  pooled-k accumulators
#define WS_VSUM   2048   // f32 [16][128]  pooled-v accumulators
#define WS_KF16   4096   // f16 [4][16][32] normalized pooled-k, A-fragment order
#define WS_VF16   5120   // f16 [4][32][16] pooled-v transposed, B-fragment order

// ---------------------------------------------------------------------------
// Zero the pooling accumulators (workspace is poisoned / stale between calls).
// ---------------------------------------------------------------------------
__global__ void zero_sums_kernel(float* __restrict__ ws) {
    int i = blockIdx.x * blockDim.x + threadIdx.x;
    if (i < 4096) ws[i] = 0.0f;
}

// ---------------------------------------------------------------------------
// Phase 1: mean-pool k/v over all 4x4 windows.
// One block per image row h (512 blocks): i = h%4 fixed. Thread layout:
//   slot = tid>>5 (0..7) -> pixels w = slot + 8t, so j = w%4 = slot%4 is FIXED
//   c4   = (tid&31)*4    -> float4 channel group; 32 lanes = 512B coalesced
// => single statically-indexed float4 accumulator per array per thread,
//    then 8 f32 atomics into the 16KB workspace accumulator.
// ---------------------------------------------------------------------------
__global__ __launch_bounds__(256) void pool_kv_kernel(const float* __restrict__ k,
                                                      const float* __restrict__ v,
                                                      float* __restrict__ ws) {
    const int h    = blockIdx.x;               // image row 0..511
    const int i    = h & 3;
    const int slot = threadIdx.x >> 5;         // 0..7
    const int c4   = (threadIdx.x & 31) << 2;  // channel base 0,4,...,124
    const int n    = (i << 2) | (slot & 3);    // window-position bucket

    const float4* krow = (const float4*)(k + (long)h * 512 * C_DIM);
    const float4* vrow = (const float4*)(v + (long)h * 512 * C_DIM);

    float4 ak = make_float4(0.f, 0.f, 0.f, 0.f);
    float4 av = make_float4(0.f, 0.f, 0.f, 0.f);
    for (int t = 0; t < 64; ++t) {
        const long idx = (long)(slot + 8 * t) * (C_DIM / 4) + (c4 >> 2);
        const float4 kk = krow[idx];
        const float4 vv = vrow[idx];
        ak.x += kk.x; ak.y += kk.y; ak.z += kk.z; ak.w += kk.w;
        av.x += vv.x; av.y += vv.y; av.z += vv.z; av.w += vv.w;
    }
    float* kd = &ws[WS_KSUM + n * C_DIM + c4];
    float* vd = &ws[WS_VSUM + n * C_DIM + c4];
    atomicAdd(kd + 0, ak.x); atomicAdd(kd + 1, ak.y);
    atomicAdd(kd + 2, ak.z); atomicAdd(kd + 3, ak.w);
    atomicAdd(vd + 0, av.x); atomicAdd(vd + 1, av.y);
    atomicAdd(vd + 2, av.z); atomicAdd(vd + 3, av.w);
}

// ---------------------------------------------------------------------------
// Phase 2: finalize. Writes the tiny pooled tables pre-converted to f16 and
// pre-ordered for direct 16B fragment loads in the attention kernel:
//   knT[h][key][ch] = l2norm(ksum/16384) per (key,head) row   (A-frag order)
//   vT [h][ch][key] = vsum/16384 transposed                   (B-frag order)
// 1 block, trivial cost, no intra-block dependencies.
// ---------------------------------------------------------------------------
__global__ void finalize_pool_kernel(float* __restrict__ ws) {
    const int t = threadIdx.x;
    const float inv = 1.0f / 16384.0f;
    _Float16* knT = (_Float16*)(ws + WS_KF16);
    _Float16* vT  = (_Float16*)(ws + WS_VF16);

    if (t < 128) {                             // vT: (h,d) per thread
        const int h = t >> 5, d = t & 31;
        for (int key = 0; key < 16; ++key) {
            const float x = ws[WS_VSUM + key * C_DIM + h * HD + d] * inv;
            vT[(h * 32 + d) * 16 + key] = (_Float16)x;
        }
    } else if (t < 192) {                      // knT: (key,h) per thread
        const int u = t - 128;                 // 0..63
        const int n = u >> 2, h = u & 3;
        const int base = n * C_DIM + h * HD;
        float ss = 0.0f;
        for (int d = 0; d < HD; ++d) {
            const float x = ws[WS_KSUM + base + d] * inv;
            ss += x * x;
        }
        const float s = 1.0f / fmaxf(sqrtf(ss), 1e-12f);
        for (int d = 0; d < HD; ++d)
            knT[(h * 16 + n) * 32 + d] = (_Float16)(ws[WS_KSUM + base + d] * inv * s);
    }
}

// ---------------------------------------------------------------------------
// Phase 3: attention. One wave32 = 16 queries x 4 heads, 3 WMMAs per head.
// Operand-swapped QK so the logits come out ALREADY TRANSPOSED:
//   QK : v_wmma_f32_16x16x32_f16  A = knT (16key x 32ch)  B = qn^T (32ch x 16q)
//        -> D[key][query]: lane (n=query, g) holds keys 8g..8g+7 in 8 VGPRs,
//           which is both the softmax-friendly layout and the exact A-fragment
//           slot order for PV. No LDS, no barriers.
//   PV : 2x v_wmma_f32_16x16x32_f16, K(keys) padded 16..31 = 0.
// knT/vT fragments are single 16B f16 loads from the precomputed tables.
// Block = 256 threads = 8 waves = 128 queries; grid = 2048 blocks.
// ---------------------------------------------------------------------------
__global__ __launch_bounds__(256) void win_attn_kernel(const float* __restrict__ q,
                                                       const float* __restrict__ mask,
                                                       const float* __restrict__ logit_scale,
                                                       const float* __restrict__ ws,
                                                       float* __restrict__ out) {
    const int lane = threadIdx.x & 31;
    const int wave = threadIdx.x >> 5;
    const int m    = lane & 15;                // A row = key; B/D column = query
    const int g    = lane >> 4;                // lane-group select
    const long q0  = ((long)blockIdx.x * 8 + wave) * 16;

    const _Float16* knT = (const _Float16*)(ws + WS_KF16);
    const _Float16* vT  = (const _Float16*)(ws + WS_VF16);

    // Mask (broadcast over heads): lane (query m, g) needs mask[q0+m][8g..8g+7]
    float mrow[8];
    const float* mk = mask + (q0 + m) * 16 + g * 8;
#pragma unroll
    for (int j = 0; j < 8; ++j) mrow[j] = mk[j];

    const float logmax = 4.60517019f;          // log(1/0.01)

    for (int h = 0; h < NH; ++h) {
        // ---- B fragment: q^T, normalized on the fly ---------------------
        // B layout (mirror of ISA A table): slot t<8 -> ch 8g+t, t>=8 -> ch 16+8g+(t-8)
        const float* qrow = q + (q0 + m) * C_DIM + (long)h * HD;
        float qv[16];
        float ss = 0.0f;
#pragma unroll
        for (int j = 0; j < 8; ++j) {
            const float x0 = qrow[8 * g + j];
            const float x1 = qrow[16 + 8 * g + j];
            qv[j] = x0; qv[8 + j] = x1;
            ss += x0 * x0 + x1 * x1;
        }
        ss += __shfl_xor(ss, 16, 32);          // join the two half-column lanes
        const float qs = 1.0f / fmaxf(sqrtf(ss), 1e-12f);
        v16h bfrag;
#pragma unroll
        for (int t = 0; t < 16; ++t) bfrag[t] = (_Float16)(qv[t] * qs);

        // ---- A fragment: knT, two direct 16B f16 loads ------------------
        const v8h alo = *(const v8h*)(knT + ((h * 16 + m) * 32 + 8 * g));
        const v8h ahi = *(const v8h*)(knT + ((h * 16 + m) * 32 + 16 + 8 * g));
        const v16h afrag = __builtin_shufflevector(alo, ahi,
            0, 1, 2, 3, 4, 5, 6, 7, 8, 9, 10, 11, 12, 13, 14, 15);

        v8f cacc = {};
        cacc = __builtin_amdgcn_wmma_f32_16x16x32_f16(false, afrag, false, bfrag,
                                                      (short)0, cacc, false, false);
        // cacc[r] = logit(query m, key r+8g) -- already transposed.

        const float sc = __expf(fminf(logit_scale[h], logmax));

        // ---- softmax over 16 keys: in-lane + one cross-half shuffle -----
        float x[8];
#pragma unroll
        for (int r = 0; r < 8; ++r) x[r] = cacc[r] * sc + mrow[r];
        float mx = x[0];
#pragma unroll
        for (int r = 1; r < 8; ++r) mx = fmaxf(mx, x[r]);
        mx = fmaxf(mx, __shfl_xor(mx, 16, 32));
        float e[8], s = 0.0f;
#pragma unroll
        for (int r = 0; r < 8; ++r) { e[r] = __expf(x[r] - mx); s += e[r]; }
        s += __shfl_xor(s, 16, 32);
        const float rinv = 1.0f / s;

        // ---- P fragment: A-layout, slot t = key 8g+t = e[t]; K 16..31 = 0
        v16h pfrag = {};
#pragma unroll
        for (int j = 0; j < 8; ++j) pfrag[j] = (_Float16)(e[j] * rinv);

        // ---- V fragments: single 16B f16 load each (keys 8g..8g+7) ------
        const v8h vlo = *(const v8h*)(vT + ((h * 32 + m) * 16 + 8 * g));
        const v8h vhi = *(const v8h*)(vT + ((h * 32 + 16 + m) * 16 + 8 * g));
        v16h vb0 = {}, vb1 = {};
#pragma unroll
        for (int j = 0; j < 8; ++j) { vb0[j] = vlo[j]; vb1[j] = vhi[j]; }

        v8f o0 = {}, o1 = {};
        o0 = __builtin_amdgcn_wmma_f32_16x16x32_f16(false, pfrag, false, vb0,
                                                    (short)0, o0, false, false);
        o1 = __builtin_amdgcn_wmma_f32_16x16x32_f16(false, pfrag, false, vb1,
                                                    (short)0, o1, false, false);

        // ---- store (D layout: lane n=hd col m; rows M=r+8g are queries)
        // o0/o1: D[query][hd]: lane holds hd column m (and m+16), query rows r+8g.
#pragma unroll
        for (int r = 0; r < 8; ++r) {
            const long row = (q0 + r + 8 * g) * C_DIM + (long)h * HD;
            out[row + m]      = o0[r];
            out[row + 16 + m] = o1[r];
        }
    }
}

// ---------------------------------------------------------------------------
// Inputs (setup_inputs order): 0=q 1=k 2=v 3=H 4=W 5=mask 6=logit_scale
// ---------------------------------------------------------------------------
extern "C" void kernel_launch(void* const* d_in, const int* in_sizes, int n_in,
                              void* d_out, int out_size, void* d_ws, size_t ws_size,
                              hipStream_t stream) {
    const float* q    = (const float*)d_in[0];
    const float* k    = (const float*)d_in[1];
    const float* v    = (const float*)d_in[2];
    const float* mask = (const float*)d_in[5];
    const float* lsc  = (const float*)d_in[6];
    float* out = (float*)d_out;
    float* ws  = (float*)d_ws;

    hipLaunchKernelGGL(zero_sums_kernel,     dim3(16),   dim3(256), 0, stream, ws);
    hipLaunchKernelGGL(pool_kv_kernel,       dim3(512),  dim3(256), 0, stream, k, v, ws);
    hipLaunchKernelGGL(finalize_pool_kernel, dim3(1),    dim3(256), 0, stream, ws);
    hipLaunchKernelGGL(win_attn_kernel,      dim3(2048), dim3(256), 0, stream,
                       q, mask, lsc, ws, out);
}
